// DeepCutModule_67413806678386
// MI455X (gfx1250) — matrix operand order repaired and checked by reading, not compiled, not run on hardware
//
#include <hip/hip_runtime.h>
#include <hip/hip_bf16.h>

// ---------------- constants (match reference) ----------------
#define NB   4
#define NT   4096
#define ND   384
#define NHH  64
#define NWW  64
#define NK   8
#define NCH  64
#define NL   4
#define NIN  386
#define KPAD 416          // 13*32, padded K for first GEMM
#define NTOK (NB*NT)      // 16384

typedef __attribute__((ext_vector_type(16))) _Float16 v16h;
typedef __attribute__((ext_vector_type(8)))  _Float16 v8h;
typedef __attribute__((ext_vector_type(8)))  float    v8f;

union AFrag { v16h v; v8h p[2]; };

__device__ __forceinline__ int degcnt1(int i) {
  int a = i < 2 ? i : 2;
  int b = (63 - i) < 2 ? (63 - i) : 2;
  return a + b + 1;
}
__device__ __forceinline__ float dis_of(int i, int j) {
  return rsqrtf((float)(degcnt1(i) * degcnt1(j)));
}

// ---------------- misc small kernels ----------------
__global__ void zero_acc_kernel(float* __restrict__ acc, int n) {
  int i = blockIdx.x * blockDim.x + threadIdx.x;
  if (i < n) acc[i] = 0.0f;
}

// Build X0 (f16): [features(384) | pos_x | pos_y | zeros..] row stride KPAD
__global__ void pack_x0_kernel(const float* __restrict__ F, _Float16* __restrict__ X0) {
  long idx = (long)blockIdx.x * blockDim.x + threadIdx.x;
  if (idx >= (long)NTOK * KPAD) return;
  int  col = (int)(idx % KPAD);
  long row = idx / KPAD;
  int  t   = (int)(row & (NT - 1));
  _Float16 v;
  if (col < ND)        v = (_Float16)F[row * ND + col];
  else if (col == ND)  v = (_Float16)((float)(t & 63) * (1.0f / 63.0f));     // pos_x
  else if (col == ND+1)v = (_Float16)((float)(t >> 6) * (1.0f / 63.0f));     // pos_y
  else                 v = (_Float16)0.0f;
  X0[idx] = v;
}

// Weight [Ksrc,N] f32 -> per-lane WMMA B-fragment layout (f16), zero pad K rows.
// dword idx = ((kblk*ntiles + tn)*32 + lane)*8 + v holds halfs (K=kblk*32+half*16+2v, n)
__global__ void pack_wfrag_kernel(const float* __restrict__ src, _Float16* __restrict__ dst,
                                  int Ksrc, int KT, int ntiles) {
  int N = ntiles * 16;
  int idx = blockIdx.x * blockDim.x + threadIdx.x;
  if (idx >= KT * ntiles * 256) return;
  int v    = idx & 7;
  int lane = (idx >> 3) & 31;
  int tn   = (idx >> 8) % ntiles;
  int kblk = idx / (256 * ntiles);
  int half = lane >> 4, l16 = lane & 15;
  int kk = kblk * 32 + half * 16 + 2 * v;
  int n  = tn * 16 + l16;
  dst[idx * 2]     = (kk     < Ksrc) ? (_Float16)src[(size_t)kk * N + n]       : (_Float16)0.0f;
  dst[idx * 2 + 1] = (kk + 1 < Ksrc) ? (_Float16)src[(size_t)(kk + 1) * N + n] : (_Float16)0.0f;
}

// ec_w1 (32,384,3,3) OIHW -> B-fragment layout per (tap, kblk(12), tn(2))
__global__ void pack_wc_frag_kernel(const float* __restrict__ w1, _Float16* __restrict__ dst) {
  int idx = blockIdx.x * blockDim.x + threadIdx.x;
  if (idx >= 9 * 12 * 2 * 256) return;
  int v    = idx & 7;
  int lane = (idx >> 3) & 31;
  int tn   = (idx >> 8) & 1;
  int kblk = (idx >> 9) % 12;
  int tap  = idx / (512 * 12);
  int ky = tap / 3, kx = tap % 3;
  int half = lane >> 4, l16 = lane & 15;
  int c = kblk * 32 + half * 16 + 2 * v;
  int o = tn * 16 + l16;
  dst[idx * 2]     = (_Float16)w1[((size_t)(o * ND + c)     * 3 + ky) * 3 + kx];
  dst[idx * 2 + 1] = (_Float16)w1[((size_t)(o * ND + c + 1) * 3 + ky) * 3 + kx];
}

// ---------------- WMMA GEMM: C[M,N] = A[M,K]f16 * Bfrag (+bias), K/N compile-time ----------------
template<int KT, int NTILE>
__global__ void gemm_wmma_kernel(const _Float16* __restrict__ A,
                                 const _Float16* __restrict__ Bfrag,
                                 const float* __restrict__ bias,
                                 float* __restrict__ C, int M) {
  constexpr int K = KT * 32;
  constexpr int N = NTILE * 16;
  int wid  = (blockIdx.x * blockDim.x + threadIdx.x) >> 5;
  int lane = threadIdx.x & 31;
  int tm = wid / NTILE, tn = wid % NTILE;
  if (tm >= (M >> 4)) return;
  int half = lane >> 4, l16 = lane & 15;
  const _Float16* arow = A + (size_t)(tm * 16 + l16) * K + half * 16;
  const _Float16* bbase = Bfrag + ((size_t)tn * 32 + lane) * 16;
  int colB = tn * 16 + l16;
  v8f acc = {};
#pragma unroll
  for (int kb = 0; kb < KT; ++kb) {
    AFrag ua;
    ua.p[0] = *(const v8h*)(arow + kb * 32);
    ua.p[1] = *(const v8h*)(arow + kb * 32 + 8);
    v16h bfr = *(const v16h*)(bbase + (size_t)kb * NTILE * 32 * 16);
    acc = __builtin_amdgcn_wmma_f32_16x16x32_f16(false, ua.v, false, bfr, (short)0, acc,
                                                 false, false);
  }
  float bv = bias ? bias[colB] : 0.0f;
#pragma unroll
  for (int v = 0; v < 8; ++v) {
    int m = tm * 16 + v + 8 * half;
    C[(size_t)m * N + colB] = acc[v] + bv;
  }
}

// ---------------- implicit-GEMM 3x3 conv 384->32 via WMMA ----------------
__global__ void conv_ec_wmma_kernel(const _Float16* __restrict__ Xh,
                                    const _Float16* __restrict__ Wfrag,
                                    const float* __restrict__ bias,
                                    float* __restrict__ E) {
  int wid  = (blockIdx.x * blockDim.x + threadIdx.x) >> 5;
  int lane = threadIdx.x & 31;
  int tn = wid & 1;      // 2 n-tiles of 16 out channels
  int mt = wid >> 1;     // B*64*4 m-tiles (16 pixels along x)
  if (mt >= NB * NHH * 4) return;
  int xt = mt & 3, y = (mt >> 2) & 63, b = mt >> 8;
  int half = lane >> 4, l16 = lane & 15;
  int px = xt * 16 + l16;      // A-row pixel x
  int oc = tn * 16 + l16;      // B-col out channel
  v8f acc = {};
  const v8h z = {};
  for (int ky = 0; ky < 3; ++ky) {
    int ys = y + ky - 1;
    if (ys < 0 || ys >= 64) continue;       // wave-uniform
    for (int kx = 0; kx < 3; ++kx) {
      int xs = px + kx - 1;
      bool ok = (xs >= 0 && xs < 64);
      int xsafe = xs < 0 ? 0 : (xs > 63 ? 63 : xs);
      const _Float16* arow = Xh + (size_t)(b * NT + ys * 64 + xsafe) * KPAD + half * 16;
      int tap = ky * 3 + kx;
      const _Float16* bbase = Wfrag + ((size_t)((tap * 12) * 2 + tn) * 32 + lane) * 16;
#pragma unroll
      for (int kb = 0; kb < 12; ++kb) {
        AFrag ua;
        v8h a0 = z, a1 = z;
        if (ok) {
          a0 = *(const v8h*)(arow + kb * 32);
          a1 = *(const v8h*)(arow + kb * 32 + 8);
        }
        ua.p[0] = a0; ua.p[1] = a1;
        v16h bfr = *(const v16h*)(bbase + (size_t)kb * 2 * 32 * 16);
        acc = __builtin_amdgcn_wmma_f32_16x16x32_f16(false, ua.v, false, bfr, (short)0, acc,
                                                     false, false);
      }
    }
  }
  float bb = bias[oc];
#pragma unroll
  for (int v = 0; v < 8; ++v) {
    int p = b * NT + y * 64 + xt * 16 + v + 8 * half;
    E[(size_t)p * 32 + oc] = acc[v] + bb;
  }
}

// ---------------- stencil SpMM: H = D^-1/2 A D^-1/2 * G (+bias), C=64 ----------------
__global__ void spmm_kernel(const float* __restrict__ G, const float* __restrict__ bias,
                            float* __restrict__ Hs) {
  long idx = (long)blockIdx.x * blockDim.x + threadIdx.x;
  if (idx >= (long)NB * NT * NCH) return;
  int c = (int)(idx & (NCH - 1));
  int t = (int)((idx >> 6) & (NT - 1));
  int b = (int)(idx >> 18);
  int i = t >> 6, j = t & 63;
  float sum = 0.0f;
  for (int dy = -2; dy <= 2; ++dy) {
    int ii = i + dy;
    if ((unsigned)ii >= 64u) continue;
    for (int dx = -2; dx <= 2; ++dx) {
      int jj = j + dx;
      if ((unsigned)jj >= 64u) continue;
      sum += dis_of(ii, jj) * G[((long)b * NT + ii * 64 + jj) * NCH + c];
    }
  }
  Hs[idx] = dis_of(i, j) * sum + (bias ? bias[c] : 0.0f);
}

// ---------------- per-token layernorm (+elu) (+residual) -> f32 (+f16) ----------------
__global__ void ln_token_kernel(const float* __restrict__ Hin, const float* __restrict__ g,
                                const float* __restrict__ bb, const float* __restrict__ prev,
                                float* __restrict__ outF, _Float16* __restrict__ outH,
                                int do_elu) {
  int tok = blockIdx.x * blockDim.x + threadIdx.x;
  if (tok >= NTOK) return;
  const float* h = Hin + (long)tok * NCH;
  float hv[NCH];
  float m = 0.0f;
#pragma unroll
  for (int c = 0; c < NCH; ++c) { hv[c] = h[c]; m += hv[c]; }
  m *= (1.0f / NCH);
  float var = 0.0f;
#pragma unroll
  for (int c = 0; c < NCH; ++c) { float d = hv[c] - m; var += d * d; }
  var *= (1.0f / NCH);
  float rs = rsqrtf(var + 1e-5f);
#pragma unroll
  for (int c = 0; c < NCH; ++c) {
    float y = (hv[c] - m) * rs * g[c] + bb[c];
    if (do_elu) y = y > 0.0f ? y : expm1f(y);
    if (prev) y += prev[(long)tok * NCH + c];
    outF[(long)tok * NCH + c] = y;
    if (outH) outH[(long)tok * NCH + c] = (_Float16)y;
  }
}

// ---------------- MLP head: logits(64->8) + softmax -> S ----------------
__global__ void mlp2_softmax_kernel(const float* __restrict__ H1, const float* __restrict__ W2,
                                    const float* __restrict__ b2, float* __restrict__ S) {
  int tok = blockIdx.x * blockDim.x + threadIdx.x;
  if (tok >= NTOK) return;
  const float* h = H1 + (long)tok * NCH;
  float lg[NK];
#pragma unroll
  for (int k = 0; k < NK; ++k) lg[k] = b2[k];
  for (int c = 0; c < NCH; ++c) {
    float hv = h[c];
#pragma unroll
    for (int k = 0; k < NK; ++k) lg[k] += hv * W2[c * NK + k];
  }
  float mx = lg[0];
#pragma unroll
  for (int k = 1; k < NK; ++k) mx = fmaxf(mx, lg[k]);
  float ssum = 0.0f;
#pragma unroll
  for (int k = 0; k < NK; ++k) { lg[k] = __expf(lg[k] - mx); ssum += lg[k]; }
  float inv = 1.0f / ssum;
#pragma unroll
  for (int k = 0; k < NK; ++k) S[(long)tok * NK + k] = lg[k] * inv;
}

// ---------------- group-norm stats over E (B,P,32), groups of 4 channels ----------------
__global__ void gn_stats_e_kernel(const float* __restrict__ E, float* __restrict__ mv) {
  __shared__ float s1[256], s2[256];
  int b = blockIdx.x >> 3, g = blockIdx.x & 7;
  float a = 0.0f, q = 0.0f;
  for (int p = threadIdx.x; p < NT; p += 256) {
    const float* e = E + ((long)b * NT + p) * 32 + g * 4;
#pragma unroll
    for (int c = 0; c < 4; ++c) { float v = e[c]; a += v; q += v * v; }
  }
  s1[threadIdx.x] = a; s2[threadIdx.x] = q;
  __syncthreads();
  for (int s = 128; s > 0; s >>= 1) {
    if ((int)threadIdx.x < s) { s1[threadIdx.x] += s1[threadIdx.x + s]; s2[threadIdx.x] += s2[threadIdx.x + s]; }
    __syncthreads();
  }
  if (threadIdx.x == 0) {
    float m = s1[0] / (4.0f * NT);
    float v = s2[0] / (4.0f * NT) - m * m;
    mv[blockIdx.x * 2]     = m;
    mv[blockIdx.x * 2 + 1] = rsqrtf(v + 1e-5f);
  }
}

// ---------------- edge weight: sigmoid(1x1 conv of relu(gn(E))) ----------------
__global__ void ew_kernel(const float* __restrict__ E, const float* __restrict__ mv,
                          const float* __restrict__ gg, const float* __restrict__ gb,
                          const float* __restrict__ w2, const float* __restrict__ b2,
                          float* __restrict__ EW) {
  int idx = blockIdx.x * blockDim.x + threadIdx.x;
  if (idx >= NB * NT) return;
  int b = idx >> 12;
  const float* e = E + (long)idx * 32;
  float acc = b2[0];
#pragma unroll
  for (int o = 0; o < 32; ++o) {
    int g = o >> 2;
    float m = mv[(b * 8 + g) * 2], rs = mv[(b * 8 + g) * 2 + 1];
    float v = (e[o] - m) * rs * gg[o] + gb[o];
    v = fmaxf(v, 0.0f);
    acc += v * w2[o];
  }
  EW[idx] = 1.0f / (1.0f + __expf(-acc));
}

// ---------------- edge-aware feature smoothness reduction ----------------
__global__ void fs_loss_kernel(const float* __restrict__ F, const float* __restrict__ EW,
                               float* __restrict__ fsacc) {
  __shared__ float sy[256], sx[256];
  int idx = blockIdx.x * blockDim.x + threadIdx.x;
  float ay = 0.0f, ax = 0.0f;
  if (idx < NB * NT) {
    int p = idx & (NT - 1);
    int y = p >> 6, x = p & 63;
    float ewv = EW[idx];
    const float4* fc = (const float4*)(F + (long)idx * ND);
    if (y > 0) {
      const float4* fu = (const float4*)(F + (long)(idx - 64) * ND);
      for (int c = 0; c < ND / 4; ++c) {
        float4 a = fc[c], bq = fu[c];
        float d0 = a.x - bq.x, d1 = a.y - bq.y, d2 = a.z - bq.z, d3 = a.w - bq.w;
        ay += d0 * d0 + d1 * d1 + d2 * d2 + d3 * d3;
      }
      ay *= ewv;
    }
    if (x > 0) {
      const float4* fl = (const float4*)(F + (long)(idx - 1) * ND);
      for (int c = 0; c < ND / 4; ++c) {
        float4 a = fc[c], bq = fl[c];
        float d0 = a.x - bq.x, d1 = a.y - bq.y, d2 = a.z - bq.z, d3 = a.w - bq.w;
        ax += d0 * d0 + d1 * d1 + d2 * d2 + d3 * d3;
      }
      ax *= ewv;
    }
  }
  sy[threadIdx.x] = ay; sx[threadIdx.x] = ax;
  __syncthreads();
  for (int s = 128; s > 0; s >>= 1) {
    if ((int)threadIdx.x < s) { sy[threadIdx.x] += sy[threadIdx.x + s]; sx[threadIdx.x] += sx[threadIdx.x + s]; }
    __syncthreads();
  }
  if (threadIdx.x == 0) {
    atomicAdd(&fsacc[0], sy[0]);
    atomicAdd(&fsacc[1], sx[0]);
  }
}

// ---------------- depthwise 3x3 conv #1 on S (B,T,K)->T1 (B,K,P) ----------------
__global__ void dwconv1_kernel(const float* __restrict__ S, const float* __restrict__ w,
                               const float* __restrict__ bias, float* __restrict__ T1) {
  int idx = blockIdx.x * blockDim.x + threadIdx.x;
  if (idx >= NB * NK * NT) return;
  int p = idx & (NT - 1);
  int k = (idx >> 12) & 7;
  int b = idx >> 15;
  int y = p >> 6, x = p & 63;
  float acc = bias[k];
  for (int ky = 0; ky < 3; ++ky) {
    int ys = y + ky - 1;
    if ((unsigned)ys >= 64u) continue;
    for (int kx = 0; kx < 3; ++kx) {
      int xs = x + kx - 1;
      if ((unsigned)xs >= 64u) continue;
      acc += w[k * 9 + ky * 3 + kx] * S[((long)b * NT + ys * 64 + xs) * NK + k];
    }
  }
  T1[idx] = acc;
}

// per-(b,k) plane mean / rstd over 4096 contiguous values
__global__ void plane_stats_kernel(const float* __restrict__ X, float* __restrict__ mv) {
  __shared__ float s1[256], s2[256];
  const float* P = X + (long)blockIdx.x * NT;
  float a = 0.0f, q = 0.0f;
  for (int p = threadIdx.x; p < NT; p += 256) { float v = P[p]; a += v; q += v * v; }
  s1[threadIdx.x] = a; s2[threadIdx.x] = q;
  __syncthreads();
  for (int s = 128; s > 0; s >>= 1) {
    if ((int)threadIdx.x < s) { s1[threadIdx.x] += s1[threadIdx.x + s]; s2[threadIdx.x] += s2[threadIdx.x + s]; }
    __syncthreads();
  }
  if (threadIdx.x == 0) {
    float m = s1[0] / NT;
    float v = s2[0] / NT - m * m;
    mv[blockIdx.x * 2] = m;
    mv[blockIdx.x * 2 + 1] = rsqrtf(v + 1e-5f);
  }
}

// depthwise conv #2 on relu(gn1(T1)) -> T2
__global__ void dwconv2_kernel(const float* __restrict__ T1, const float* __restrict__ mv1,
                               const float* __restrict__ g1, const float* __restrict__ b1,
                               const float* __restrict__ w, const float* __restrict__ bias,
                               float* __restrict__ T2) {
  int idx = blockIdx.x * blockDim.x + threadIdx.x;
  if (idx >= NB * NK * NT) return;
  int p = idx & (NT - 1);
  int plane = idx >> 12;       // b*8 + k
  int k = plane & 7;
  int y = p >> 6, x = p & 63;
  float m = mv1[plane * 2], rs = mv1[plane * 2 + 1];
  float gg = g1[k], bbv = b1[k];
  const float* P = T1 + (long)plane * NT;
  float acc = bias[k];
  for (int ky = 0; ky < 3; ++ky) {
    int ys = y + ky - 1;
    if ((unsigned)ys >= 64u) continue;
    for (int kx = 0; kx < 3; ++kx) {
      int xs = x + kx - 1;
      if ((unsigned)xs >= 64u) continue;
      float v = (P[ys * 64 + xs] - m) * rs * gg + bbv;
      v = fmaxf(v, 0.0f);
      acc += w[k * 9 + ky * 3 + kx] * v;
    }
  }
  T2[idx] = acc;
}

// sm_loss[b] partial: sum over (k,p) of (gn2(T2) - S)^2
__global__ void sm_loss_kernel(const float* __restrict__ T2, const float* __restrict__ mv2,
                               const float* __restrict__ g2, const float* __restrict__ b2,
                               const float* __restrict__ S, float* __restrict__ smacc) {
  __shared__ float s1[256];
  int plane = blockIdx.x;
  int b = plane >> 3, k = plane & 7;
  float m = mv2[plane * 2], rs = mv2[plane * 2 + 1];
  float gg = g2[k], bbv = b2[k];
  const float* P = T2 + (long)plane * NT;
  float a = 0.0f;
  for (int p = threadIdx.x; p < NT; p += 256) {
    float v = (P[p] - m) * rs * gg + bbv;
    float d = v - S[((long)b * NT + p) * NK + k];
    a += d * d;
  }
  s1[threadIdx.x] = a;
  __syncthreads();
  for (int s = 128; s > 0; s >>= 1) {
    if ((int)threadIdx.x < s) s1[threadIdx.x] += s1[threadIdx.x + s];
    __syncthreads();
  }
  if (threadIdx.x == 0) atomicAdd(&smacc[b], s1[0]);
}

// normalized-cut num/den + S^T S accumulation
__global__ void cut_loss_kernel(const float* __restrict__ S, float* __restrict__ numacc,
                                float* __restrict__ denacc, float* __restrict__ ssacc) {
  __shared__ float ssh[64];
  __shared__ float red[256];
  int b  = blockIdx.x >> 4;
  int t  = ((blockIdx.x & 15) << 8) + threadIdx.x;
  if (threadIdx.x < 64) ssh[threadIdx.x] = 0.0f;
  __syncthreads();
  int i = t >> 6, j = t & 63;
  const float* Sb = S + (long)b * NT * NK;
  float st[NK], as[NK];
#pragma unroll
  for (int k = 0; k < NK; ++k) { st[k] = Sb[(long)t * NK + k]; as[k] = 0.0f; }
  float dsum = 0.0f;
  for (int dy = -2; dy <= 2; ++dy) {
    int ii = i + dy;
    if ((unsigned)ii >= 64u) continue;
    for (int dx = -2; dx <= 2; ++dx) {
      int jj = j + dx;
      if ((unsigned)jj >= 64u) continue;
      float ds = dis_of(ii, jj);
      dsum += ds;
      const float* Ss = Sb + (long)(ii * 64 + jj) * NK;
#pragma unroll
      for (int k = 0; k < NK; ++k) as[k] += ds * Ss[k];
    }
  }
  float dt = dis_of(i, j);
  float numv = 0.0f, s2 = 0.0f;
#pragma unroll
  for (int k = 0; k < NK; ++k) { numv += st[k] * dt * as[k]; s2 += st[k] * st[k]; }
  float denv = (dt * dsum) * s2;     // degA[t] * sum_k S^2
#pragma unroll
  for (int k = 0; k < NK; ++k)
#pragma unroll
    for (int l = 0; l < NK; ++l)
      atomicAdd(&ssh[k * NK + l], st[k] * st[l]);
  red[threadIdx.x] = numv;
  __syncthreads();
  for (int s = 128; s > 0; s >>= 1) {
    if ((int)threadIdx.x < s) red[threadIdx.x] += red[threadIdx.x + s];
    __syncthreads();
  }
  if (threadIdx.x == 0) atomicAdd(&numacc[b], red[0]);
  __syncthreads();
  red[threadIdx.x] = denv;
  __syncthreads();
  for (int s = 128; s > 0; s >>= 1) {
    if ((int)threadIdx.x < s) red[threadIdx.x] += red[threadIdx.x + s];
    __syncthreads();
  }
  if (threadIdx.x == 0) atomicAdd(&denacc[b], red[0]);
  __syncthreads();
  if (threadIdx.x < 64) atomicAdd(&ssacc[b * 64 + threadIdx.x], ssh[threadIdx.x]);
}

// final scalar loss
__global__ void finalize_kernel(const float* __restrict__ numacc, const float* __restrict__ denacc,
                                const float* __restrict__ ssacc, const float* __restrict__ smacc,
                                const float* __restrict__ fsacc, float* __restrict__ out) {
  if (threadIdx.x != 0 || blockIdx.x != 0) return;
  float lsum = 0.0f;
  float tgt = 1.0f / (sqrtf(8.0f) + 1e-6f);
  for (int b = 0; b < NB; ++b) {
    float loss_c = -numacc[b] / (denacc[b] + 1e-6f);
    float ssn = 0.0f;
    for (int i = 0; i < 64; ++i) { float v = ssacc[b * 64 + i]; ssn += v * v; }
    ssn = sqrtf(ssn);
    float dsum = 0.0f;
    for (int i = 0; i < 8; ++i)
      for (int j = 0; j < 8; ++j) {
        float v = ssacc[b * 64 + i * 8 + j] / (ssn + 1e-6f) - ((i == j) ? tgt : 0.0f);
        dsum += v * v;
      }
    float loss_o = sqrtf(dsum);
    float smb = smacc[b] / (8.0f * 4096.0f);
    lsum += loss_c + loss_o + 0.01f * smb;
  }
  float loss_y = fsacc[0] / (4.0f * 384.0f * 63.0f * 64.0f) / 384.0f;
  float loss_x = fsacc[1] / (4.0f * 384.0f * 64.0f * 63.0f) / 384.0f;
  float feat_sm = 0.5f * (loss_y + loss_x);
  out[NTOK * NK] = lsum / 4.0f + 0.09f * feat_sm;
}

// =================================================================
extern "C" void kernel_launch(void* const* d_in, const int* in_sizes, int n_in,
                              void* d_out, int out_size, void* d_ws, size_t ws_size,
                              hipStream_t stream) {
  const float* tf       = (const float*)d_in[0];
  const float* gcn_w0   = (const float*)d_in[1];
  const float* gcn_b0   = (const float*)d_in[2];
  const float* gcn_w    = (const float*)d_in[3];
  const float* gcn_b    = (const float*)d_in[4];
  const float* ln_g     = (const float*)d_in[5];
  const float* ln_b     = (const float*)d_in[6];
  const float* res_w    = (const float*)d_in[7];
  const float* res_b    = (const float*)d_in[8];
  const float* res_ln_g = (const float*)d_in[9];
  const float* res_ln_b = (const float*)d_in[10];
  const float* mlp_w1   = (const float*)d_in[11];
  const float* mlp_b1   = (const float*)d_in[12];
  const float* mlp_ln_g = (const float*)d_in[13];
  const float* mlp_ln_b = (const float*)d_in[14];
  const float* mlp_w2   = (const float*)d_in[15];
  const float* mlp_b2   = (const float*)d_in[16];
  const float* sm_w1    = (const float*)d_in[17];
  const float* sm_b1    = (const float*)d_in[18];
  const float* sm_g1g   = (const float*)d_in[19];
  const float* sm_g1b   = (const float*)d_in[20];
  const float* sm_w2    = (const float*)d_in[21];
  const float* sm_b2    = (const float*)d_in[22];
  const float* sm_g2g   = (const float*)d_in[23];
  const float* sm_g2b   = (const float*)d_in[24];
  const float* ec_w1    = (const float*)d_in[25];
  const float* ec_b1    = (const float*)d_in[26];
  const float* ec_gn_g  = (const float*)d_in[27];
  const float* ec_gn_b  = (const float*)d_in[28];
  const float* ec_w2    = (const float*)d_in[29];
  const float* ec_b2    = (const float*)d_in[30];
  (void)in_sizes; (void)n_in; (void)out_size; (void)ws_size;

  float* Sout = (float*)d_out;   // (B,T,K) then loss at [131072]

  // ---- workspace layout ----
  char* ws = (char*)d_ws;
  size_t off = 0;
  auto alloc = [&](size_t bytes) -> char* {
    char* p = ws + off;
    off += (bytes + 255) & ~(size_t)255;
    return p;
  };
  _Float16* X0H  = (_Float16*)alloc((size_t)NTOK * KPAD * 2);
  _Float16* W0H  = (_Float16*)alloc((size_t)13 * 4 * 256 * 2 * 2);   // KT=13, ntiles=4
  _Float16* RWH  = (_Float16*)alloc((size_t)13 * 4 * 256 * 2 * 2);
  _Float16* WLH  = (_Float16*)alloc((size_t)3 * 2 * 4 * 256 * 2 * 2); // 3 layers, KT=2
  _Float16* MW1H = (_Float16*)alloc((size_t)2 * 4 * 256 * 2 * 2);
  _Float16* WCH  = (_Float16*)alloc((size_t)9 * 12 * 2 * 256 * 2 * 2);
  float* G   = (float*)alloc((size_t)NTOK * NCH * 4);
  float* HS  = (float*)alloc((size_t)NTOK * NCH * 4);
  float* RES = (float*)alloc((size_t)NTOK * NCH * 4);
  float* XF  = (float*)alloc((size_t)NTOK * NCH * 4);
  _Float16* XH = (_Float16*)alloc((size_t)NTOK * NCH * 2);
  float* E   = (float*)alloc((size_t)NB * NT * 32 * 4);
  float* EW  = (float*)alloc((size_t)NB * NT * 4);
  float* T1  = (float*)alloc((size_t)NB * NK * NT * 4);
  float* T2  = (float*)alloc((size_t)NB * NK * NT * 4);
  float* EMV = (float*)alloc(64 * 4);
  float* MV1 = (float*)alloc(64 * 4);
  float* MV2 = (float*)alloc(64 * 4);
  float* ACC = (float*)alloc(512 * 4);
  float* numacc = ACC;         // 4
  float* denacc = ACC + 4;     // 4
  float* smacc  = ACC + 8;     // 4
  float* fsacc  = ACC + 12;    // 2
  float* ssacc  = ACC + 16;    // 256

  const int LFRAG = 2 * 4 * 256 * 2;   // halfs per 64x64 layer-weight fragment block

  // ---- zero accumulators (graph-replay safe) ----
  zero_acc_kernel<<<2, 256, 0, stream>>>(ACC, 512);

  // ---- pack f16 operands ----
  pack_x0_kernel<<<(NTOK * KPAD + 255) / 256, 256, 0, stream>>>(tf, X0H);
  pack_wfrag_kernel<<<(13 * 4 * 256 + 255) / 256, 256, 0, stream>>>(gcn_w0, W0H, NIN, 13, 4);
  pack_wfrag_kernel<<<(13 * 4 * 256 + 255) / 256, 256, 0, stream>>>(res_w, RWH, NIN, 13, 4);
  for (int i = 0; i < 3; ++i)
    pack_wfrag_kernel<<<(2 * 4 * 256 + 255) / 256, 256, 0, stream>>>(
        gcn_w + (size_t)i * NCH * NCH, WLH + (size_t)i * LFRAG, NCH, 2, 4);
  pack_wfrag_kernel<<<(2 * 4 * 256 + 255) / 256, 256, 0, stream>>>(mlp_w1, MW1H, NCH, 2, 4);
  pack_wc_frag_kernel<<<(9 * 12 * 2 * 256 + 255) / 256, 256, 0, stream>>>(ec_w1, WCH);

  const int gemmTiles64 = (NTOK / 16) * (NCH / 16);              // 4096 waves
  const int gemmBlocks  = (gemmTiles64 * 32 + 255) / 256;        // 512

  // ---- edge-conv branch (feature smoothness) ----
  conv_ec_wmma_kernel<<<(NB * NHH * 4 * 2 * 32 + 255) / 256, 256, 0, stream>>>(X0H, WCH, ec_b1, E);
  gn_stats_e_kernel<<<NB * 8, 256, 0, stream>>>(E, EMV);
  ew_kernel<<<(NB * NT + 255) / 256, 256, 0, stream>>>(E, EMV, ec_gn_g, ec_gn_b, ec_w2, ec_b2, EW);
  fs_loss_kernel<<<(NB * NT + 255) / 256, 256, 0, stream>>>(tf, EW, fsacc);

  // ---- GNN layer 0 ----
  gemm_wmma_kernel<13, 4><<<gemmBlocks, 256, 0, stream>>>(X0H, W0H, nullptr, G, NTOK);
  gemm_wmma_kernel<13, 4><<<gemmBlocks, 256, 0, stream>>>(X0H, RWH, res_b, RES, NTOK);
  ln_token_kernel<<<(NTOK + 255) / 256, 256, 0, stream>>>(RES, res_ln_g, res_ln_b, nullptr,
                                                          RES, nullptr, 0);
  spmm_kernel<<<(int)(((long)NB * NT * NCH + 255) / 256), 256, 0, stream>>>(G, gcn_b0, HS);
  ln_token_kernel<<<(NTOK + 255) / 256, 256, 0, stream>>>(HS, ln_g, ln_b, RES, XF, XH, 1);

  // ---- GNN layers 1..3 ----
  for (int i = 1; i < NL; ++i) {
    gemm_wmma_kernel<2, 4><<<gemmBlocks, 256, 0, stream>>>(XH, WLH + (size_t)(i - 1) * LFRAG,
                                                           nullptr, G, NTOK);
    spmm_kernel<<<(int)(((long)NB * NT * NCH + 255) / 256), 256, 0, stream>>>(G, gcn_b + (i - 1) * NCH, HS);
    ln_token_kernel<<<(NTOK + 255) / 256, 256, 0, stream>>>(HS, ln_g + i * NCH, ln_b + i * NCH,
                                                            XF, XF, XH, 1);
  }

  // ---- MLP head -> S (softmax) ----
  gemm_wmma_kernel<2, 4><<<gemmBlocks, 256, 0, stream>>>(XH, MW1H, mlp_b1, G, NTOK);
  ln_token_kernel<<<(NTOK + 255) / 256, 256, 0, stream>>>(G, mlp_ln_g, mlp_ln_b, nullptr,
                                                          HS, nullptr, 1);
  mlp2_softmax_kernel<<<(NTOK + 255) / 256, 256, 0, stream>>>(HS, mlp_w2, mlp_b2, Sout);

  // ---- spatial smoothness ----
  dwconv1_kernel<<<(NB * NK * NT + 255) / 256, 256, 0, stream>>>(Sout, sm_w1, sm_b1, T1);
  plane_stats_kernel<<<NB * NK, 256, 0, stream>>>(T1, MV1);
  dwconv2_kernel<<<(NB * NK * NT + 255) / 256, 256, 0, stream>>>(T1, MV1, sm_g1g, sm_g1b,
                                                                 sm_w2, sm_b2, T2);
  plane_stats_kernel<<<NB * NK, 256, 0, stream>>>(T2, MV2);
  sm_loss_kernel<<<NB * NK, 256, 0, stream>>>(T2, MV2, sm_g2g, sm_g2b, Sout, smacc);

  // ---- normalized-cut + orthogonality ----
  cut_loss_kernel<<<NB * 16, 256, 0, stream>>>(Sout, numacc, denacc, ssacc);

  // ---- combine ----
  finalize_kernel<<<1, 64, 0, stream>>>(numacc, denacc, ssacc, smacc, fsacc, Sout);
}
